// MultiHeadAttention_867583394604
// MI455X (gfx1250) — matrix operand order repaired and checked by reading, not compiled, not run on hardware
//
#include <hip/hip_runtime.h>
#include <hip/hip_bf16.h>

// ---------------------------------------------------------------------------
// MHA forward for MI455X (gfx1250): bf16 WMMA everywhere, fp32 accumulate.
//   pass 1-3: Q/K/V = x @ W + b   (fp32 in -> bf16 split-head [B,H,S,Dh])
//   pass 4  : flash attention per (b,h), online softmax, bf16 P/V WMMA
//   pass 5  : out = attn @ Wo + bo (bf16 A -> fp32 out)
// Register double-buffering hides HBM latency behind the WMMA pipes.
// ---------------------------------------------------------------------------

#define D_MODEL   1024
#define NUM_HEADS 16
#define HEAD_DIM  64
#define BATCH     4
#define SEQ       2048

typedef __attribute__((ext_vector_type(16))) __bf16 v16bf;
typedef __attribute__((ext_vector_type(8)))  float  v8f;

union FragBF { v16bf v; unsigned int u[8]; };

__device__ __forceinline__ unsigned short f2bf(float f) {
    unsigned int u = __builtin_bit_cast(unsigned int, f);
    unsigned int r = u + 0x7FFFu + ((u >> 16) & 1u);   // round-to-nearest-even
    return (unsigned short)(r >> 16);
}
__device__ __forceinline__ unsigned int pack2bf(float lo, float hi) {
    return (unsigned int)f2bf(lo) | ((unsigned int)f2bf(hi) << 16);
}

// ---------------------------------------------------------------------------
// Tiled bf16 WMMA GEMM: C[M,N] = A[M,K] * W[K,N] + bias
// Block = 256 threads (8 waves), tile 128x128, K-step 32.
// Waves 2(M) x 4(N); each wave owns 4x2 16x16 f32 accumulators (8 WMMA/iter).
// Next K-tile is loaded into registers while the current tile computes.
// ---------------------------------------------------------------------------
template<bool A_BF16, bool SPLIT_OUT>
__global__ __launch_bounds__(256, 2) void gemm_bf16_wmma(
    const void* __restrict__ Aptr, const float* __restrict__ W,
    const float* __restrict__ bias, void* __restrict__ Out,
    int M, int N, int K)
{
    __shared__ unsigned int   lds_a[128][17];   // [row][kpair], +1 pad
    __shared__ unsigned short lds_b[128][34];   // [n][k], +2 pad (pair reads)

    const int tid   = threadIdx.x;
    const int lane  = tid & 31;
    const int wave  = tid >> 5;
    const int waveM = wave >> 2;      // 0..1
    const int waveN = wave & 3;       // 0..3
    const int l16   = lane & 15;
    const int khalf = lane >> 4;

    const int m0 = blockIdx.y * 128;
    const int n0 = blockIdx.x * 128;

    v8f acc[4][2];
    #pragma unroll
    for (int i = 0; i < 4; ++i)
        #pragma unroll
        for (int j = 0; j < 2; ++j) { v8f z = {}; acc[i][j] = z; }

    const int arow  = tid >> 1;       // 0..127
    const int ahalf = tid & 1;        // which 16 of the 32 K columns
    const int bk    = tid >> 3;       // 0..31
    const int bn0   = (tid & 7) * 16; // 0..112

    // register staging for the incoming tile
    unsigned int aU[8];               // A as bf16 pairs (A_BF16 path)
    float        aF[16];              // A as fp32      (convert path)
    float        wF[16];              // W slice (always fp32 in)

    // ---- load tile k0 = 0 into registers ----
    {
        if (A_BF16) {
            const unsigned int* src = (const unsigned int*)Aptr +
                (((size_t)(m0 + arow) * K) >> 1) + ahalf * 8;
            #pragma unroll
            for (int j = 0; j < 8; ++j) aU[j] = src[j];
        } else {
            const float* src = (const float*)Aptr +
                (size_t)(m0 + arow) * K + ahalf * 16;
            #pragma unroll
            for (int j = 0; j < 16; ++j) aF[j] = src[j];
        }
        const float* srcW = W + (size_t)bk * N + n0 + bn0;
        #pragma unroll
        for (int j = 0; j < 16; ++j) wF[j] = srcW[j];
    }

    for (int k0 = 0; k0 < K; k0 += 32) {
        __syncthreads();
        // ---- commit staged registers to LDS (bf16) ----
        if (A_BF16) {
            #pragma unroll
            for (int j = 0; j < 8; ++j) lds_a[arow][ahalf * 8 + j] = aU[j];
        } else {
            #pragma unroll
            for (int j = 0; j < 8; ++j)
                lds_a[arow][ahalf * 8 + j] = pack2bf(aF[2 * j], aF[2 * j + 1]);
        }
        #pragma unroll
        for (int j = 0; j < 16; ++j)
            lds_b[bn0 + j][bk] = f2bf(wF[j]);
        __syncthreads();

        // ---- issue next tile's global loads (overlap with WMMA below) ----
        const int kn = k0 + 32;
        if (kn < K) {
            if (A_BF16) {
                const unsigned int* src = (const unsigned int*)Aptr +
                    (((size_t)(m0 + arow) * K + kn) >> 1) + ahalf * 8;
                #pragma unroll
                for (int j = 0; j < 8; ++j) aU[j] = src[j];
                if (kn + 32 < K)
                    __builtin_prefetch((const char*)Aptr +
                        ((size_t)(m0 + arow) * K + kn + 32) * 2, 0, 0);
            } else {
                const float* src = (const float*)Aptr +
                    (size_t)(m0 + arow) * K + kn + ahalf * 16;
                #pragma unroll
                for (int j = 0; j < 16; ++j) aF[j] = src[j];
                if (kn + 32 < K)
                    __builtin_prefetch((const char*)Aptr +
                        ((size_t)(m0 + arow) * K + kn + 32) * 4, 0, 0);
            }
            const float* srcW = W + (size_t)(kn + bk) * N + n0 + bn0;
            #pragma unroll
            for (int j = 0; j < 16; ++j) wF[j] = srcW[j];
        }

        // ---- fragment loads (ISA A/B 16-bit layouts) + WMMA ----
        FragBF af[4], bfrg[2];
        #pragma unroll
        for (int mi = 0; mi < 4; ++mi) {
            int row = waveM * 64 + mi * 16 + l16;
            #pragma unroll
            for (int v = 0; v < 8; ++v) {
                int p = (v < 4 ? v : v + 4) + khalf * 4;   // A pair pattern
                af[mi].u[v] = lds_a[row][p];
            }
        }
        #pragma unroll
        for (int ni = 0; ni < 2; ++ni) {
            int col = waveN * 32 + ni * 16 + l16;
            #pragma unroll
            for (int v = 0; v < 8; ++v) {
                int p = (v < 4 ? v : v + 4) + khalf * 4;   // B pair pattern
                bfrg[ni].u[v] = *(const unsigned int*)&lds_b[col][2 * p];
            }
        }
        #pragma unroll
        for (int mi = 0; mi < 4; ++mi)
            #pragma unroll
            for (int ni = 0; ni < 2; ++ni)
                acc[mi][ni] = __builtin_amdgcn_wmma_f32_16x16x32_bf16(
                    false, af[mi].v, false, bfrg[ni].v,
                    (short)0, acc[mi][ni], false, false);
    }

    // ---- epilogue: bias + store ----
    #pragma unroll
    for (int ni = 0; ni < 2; ++ni) {
        int gn = n0 + waveN * 32 + ni * 16 + l16;
        float bv = bias[gn];
        #pragma unroll
        for (int mi = 0; mi < 4; ++mi) {
            #pragma unroll
            for (int v = 0; v < 8; ++v) {
                int gm = m0 + waveM * 64 + mi * 16 + v + 8 * khalf;
                float val = acc[mi][ni][v] + bv;
                if (SPLIT_OUT) {
                    int b = gm >> 11, s = gm & (SEQ - 1);
                    int h = gn >> 6,  d = gn & (HEAD_DIM - 1);
                    size_t idx = (((size_t)b * NUM_HEADS + h) * SEQ + s) * HEAD_DIM + d;
                    ((unsigned short*)Out)[idx] = f2bf(val);
                } else {
                    ((float*)Out)[(size_t)gm * N + gn] = val;
                }
            }
        }
    }
}

// ---------------------------------------------------------------------------
// Flash attention: grid (S/128, H, B), 256 threads = 8 waves.
// Each wave owns 16 query rows; streams 64-key tiles of K/V through LDS.
// K and V are both fetched with coalesced dword copies; the key-major
// transpose of V happens LDS->LDS. Next tile prefetched into registers.
// ---------------------------------------------------------------------------
__global__ __launch_bounds__(256, 2) void flash_attn_wmma(
    const unsigned short* __restrict__ Q,
    const unsigned short* __restrict__ Kb,
    const unsigned short* __restrict__ Vb,
    unsigned short* __restrict__ Attn)
{
    __shared__ unsigned short lds_k [64][66];     // [key][d]
    __shared__ unsigned short lds_v [64][66];     // [key][d] (staging)
    __shared__ unsigned short lds_vt[64][66];     // [d][key] (transposed)
    __shared__ unsigned short lds_p [8][16][66];  // per-wave P staging

    const int tid   = threadIdx.x;
    const int lane  = tid & 31;
    const int wave  = tid >> 5;
    const int l16   = lane & 15;
    const int khalf = lane >> 4;

    const int h  = blockIdx.y;
    const int b  = blockIdx.z;
    const int q0 = blockIdx.x * 128 + wave * 16;
    const size_t bh = (size_t)b * NUM_HEADS + h;

    // Q fragments stay in registers (A-layout, K-dim = d, two 32-wide chunks)
    FragBF qa[2];
    {
        const unsigned int* qrow =
            (const unsigned int*)(Q + (bh * SEQ + q0 + l16) * HEAD_DIM);
        #pragma unroll
        for (int f = 0; f < 2; ++f)
            #pragma unroll
            for (int v = 0; v < 8; ++v) {
                int p = (v < 4 ? v : v + 4) + khalf * 4;
                qa[f].u[v] = qrow[f * 16 + p];
            }
    }

    v8f oacc[4];
    #pragma unroll
    for (int dt = 0; dt < 4; ++dt) { v8f z = {}; oacc[dt] = z; }
    float m[8], l[8];
    #pragma unroll
    for (int v = 0; v < 8; ++v) { m[v] = -1e30f; l[v] = 0.0f; }

    const float sscale = 0.125f * 1.44269504f;   // 1/sqrt(64) * log2(e)

    // per-thread coalesced copy slots: 2048 dwords per 64x64 bf16 tile
    const int cp_key = (tid * 8) >> 5;            // 8 dwords/thread, key row
    const int cp_dp0 = (tid * 8) & 31;            // first d-pair (always 8-aligned... safe: tid*8 % 32)

    unsigned int kreg[8], vreg[8];
    {   // tile 0
        const unsigned int* sk = (const unsigned int*)(Kb + (bh * SEQ) * HEAD_DIM);
        const unsigned int* sv = (const unsigned int*)(Vb + (bh * SEQ) * HEAD_DIM);
        #pragma unroll
        for (int j = 0; j < 8; ++j) {
            kreg[j] = sk[cp_key * 32 + cp_dp0 + j];
            vreg[j] = sv[cp_key * 32 + cp_dp0 + j];
        }
    }

    for (int kt = 0; kt < SEQ; kt += 64) {
        __syncthreads();
        // ---- commit staged K/V registers to LDS ----
        #pragma unroll
        for (int j = 0; j < 8; ++j) {
            *(unsigned int*)&lds_k[cp_key][2 * (cp_dp0 + j)] = kreg[j];
            *(unsigned int*)&lds_v[cp_key][2 * (cp_dp0 + j)] = vreg[j];
        }
        __syncthreads();

        // ---- transpose V LDS->LDS: lds_vt[d][key], pairs along key ----
        #pragma unroll
        for (int j = 0; j < 8; ++j) {
            int idx = tid + j * 256;              // 0..2047
            int d = idx & 63, kp = idx >> 6;      // kp = key pair 0..31
            unsigned int lo = lds_v[2 * kp][d];
            unsigned int hi = lds_v[2 * kp + 1][d];
            *(unsigned int*)&lds_vt[d][2 * kp] = lo | (hi << 16);
        }
        // ---- issue next tile's global loads (overlap with WMMA below) ----
        if (kt + 64 < SEQ) {
            const unsigned int* sk =
                (const unsigned int*)(Kb + (bh * SEQ + kt + 64) * HEAD_DIM);
            const unsigned int* sv =
                (const unsigned int*)(Vb + (bh * SEQ + kt + 64) * HEAD_DIM);
            #pragma unroll
            for (int j = 0; j < 8; ++j) {
                kreg[j] = sk[cp_key * 32 + cp_dp0 + j];
                vreg[j] = sv[cp_key * 32 + cp_dp0 + j];
            }
        }
        __syncthreads();

        // ---- scores S[16 x 64] = Q @ K^T, four 16x16 tiles ----
        v8f st[4];
        #pragma unroll
        for (int nt = 0; nt < 4; ++nt) {
            FragBF kf0, kf1;
            int key = nt * 16 + l16;
            #pragma unroll
            for (int v = 0; v < 8; ++v) {
                int p = (v < 4 ? v : v + 4) + khalf * 4;
                kf0.u[v] = *(const unsigned int*)&lds_k[key][2 * p];
                kf1.u[v] = *(const unsigned int*)&lds_k[key][2 * (16 + p)];
            }
            v8f z = {};
            z = __builtin_amdgcn_wmma_f32_16x16x32_bf16(
                    false, qa[0].v, false, kf0.v, (short)0, z, false, false);
            z = __builtin_amdgcn_wmma_f32_16x16x32_bf16(
                    false, qa[1].v, false, kf1.v, (short)0, z, false, false);
            st[nt] = z;
        }
        #pragma unroll
        for (int nt = 0; nt < 4; ++nt)
            #pragma unroll
            for (int v = 0; v < 8; ++v) st[nt][v] *= sscale;

        // ---- online softmax (log2 domain); row = (v, lane-half) ----
        float mn[8], corr[8], rl[8];
        #pragma unroll
        for (int v = 0; v < 8; ++v) {
            float r = fmaxf(fmaxf(st[0][v], st[1][v]), fmaxf(st[2][v], st[3][v]));
            #pragma unroll
            for (int sh = 1; sh < 16; sh <<= 1)
                r = fmaxf(r, __shfl_xor(r, sh, 32));
            mn[v]   = fmaxf(m[v], r);
            corr[v] = exp2f(m[v] - mn[v]);
            rl[v]   = 0.0f;
        }
        #pragma unroll
        for (int nt = 0; nt < 4; ++nt)
            #pragma unroll
            for (int v = 0; v < 8; ++v) {
                float pv = exp2f(st[nt][v] - mn[v]);
                rl[v] += pv;
                lds_p[wave][v + 8 * khalf][nt * 16 + l16] = f2bf(pv);
            }
        #pragma unroll
        for (int v = 0; v < 8; ++v) {
            float r = rl[v];
            #pragma unroll
            for (int sh = 1; sh < 16; sh <<= 1) r += __shfl_xor(r, sh, 32);
            l[v] = l[v] * corr[v] + r;
            m[v] = mn[v];
        }
        #pragma unroll
        for (int dt = 0; dt < 4; ++dt)
            #pragma unroll
            for (int v = 0; v < 8; ++v) oacc[dt][v] *= corr[v];

        __syncthreads();   // P staged in LDS visible / DS ordering

        // ---- O += P[16x64] @ V[64x64] ----
        FragBF pf[2];
        #pragma unroll
        for (int f = 0; f < 2; ++f)
            #pragma unroll
            for (int v = 0; v < 8; ++v) {
                int p = (v < 4 ? v : v + 4) + khalf * 4;
                pf[f].u[v] =
                    *(const unsigned int*)&lds_p[wave][l16][2 * (f * 16 + p)];
            }
        #pragma unroll
        for (int dt = 0; dt < 4; ++dt) {
            FragBF vf0, vf1;
            int d = dt * 16 + l16;
            #pragma unroll
            for (int v = 0; v < 8; ++v) {
                int p = (v < 4 ? v : v + 4) + khalf * 4;
                vf0.u[v] = *(const unsigned int*)&lds_vt[d][2 * p];
                vf1.u[v] = *(const unsigned int*)&lds_vt[d][2 * (16 + p)];
            }
            oacc[dt] = __builtin_amdgcn_wmma_f32_16x16x32_bf16(
                false, pf[0].v, false, vf0.v, (short)0, oacc[dt], false, false);
            oacc[dt] = __builtin_amdgcn_wmma_f32_16x16x32_bf16(
                false, pf[1].v, false, vf1.v, (short)0, oacc[dt], false, false);
        }
    }

    // ---- normalize and write merged-head bf16 [B, S, H*Dh] ----
    float inv[8];
    #pragma unroll
    for (int v = 0; v < 8; ++v) inv[v] = 1.0f / l[v];
    #pragma unroll
    for (int dt = 0; dt < 4; ++dt)
        #pragma unroll
        for (int v = 0; v < 8; ++v) {
            int row = q0 + v + 8 * khalf;
            int col = h * HEAD_DIM + dt * 16 + l16;
            Attn[((size_t)b * SEQ + row) * D_MODEL + col] =
                f2bf(oacc[dt][v] * inv[v]);
        }
}

// ---------------------------------------------------------------------------
extern "C" void kernel_launch(void* const* d_in, const int* in_sizes, int n_in,
                              void* d_out, int out_size, void* d_ws, size_t ws_size,
                              hipStream_t stream)
{
    (void)in_sizes; (void)n_in; (void)out_size; (void)ws_size;

    const float* query = (const float*)d_in[0];
    const float* key_i = (const float*)d_in[1];
    const float* value = (const float*)d_in[2];
    const float* Wq = (const float*)d_in[3];
    const float* bq = (const float*)d_in[4];
    const float* Wk = (const float*)d_in[5];
    const float* bk = (const float*)d_in[6];
    const float* Wv = (const float*)d_in[7];
    const float* bv = (const float*)d_in[8];
    const float* Wo = (const float*)d_in[9];
    const float* bo = (const float*)d_in[10];

    const size_t qkv_elems = (size_t)BATCH * NUM_HEADS * SEQ * HEAD_DIM; // 8M
    unsigned short* Qb  = (unsigned short*)d_ws;
    unsigned short* Kbf = Qb  + qkv_elems;
    unsigned short* Vbf = Kbf + qkv_elems;
    unsigned short* Ab  = Vbf + qkv_elems;   // attn bf16 [B,S,D]

    const int M = BATCH * SEQ;               // 8192
    dim3 blk(256);
    dim3 ggrid(D_MODEL / 128, M / 128);      // (8, 64)

    gemm_bf16_wmma<false, true><<<ggrid, blk, 0, stream>>>(
        query, Wq, bq, Qb,  M, D_MODEL, D_MODEL);
    gemm_bf16_wmma<false, true><<<ggrid, blk, 0, stream>>>(
        key_i, Wk, bk, Kbf, M, D_MODEL, D_MODEL);
    gemm_bf16_wmma<false, true><<<ggrid, blk, 0, stream>>>(
        value, Wv, bv, Vbf, M, D_MODEL, D_MODEL);

    dim3 fgrid(SEQ / 128, NUM_HEADS, BATCH); // (16, 16, 4)
    flash_attn_wmma<<<fgrid, blk, 0, stream>>>(Qb, Kbf, Vbf, Ab);

    gemm_bf16_wmma<true, false><<<ggrid, blk, 0, stream>>>(
        Ab, Wo, bo, d_out, M, D_MODEL, D_MODEL);
}
